// TemporalGATTransformer_33612414059029
// MI455X (gfx1250) — matrix operand order repaired and checked by reading, not compiled, not run on hardware
//
#include <hip/hip_runtime.h>
#include <hip/hip_bf16.h>

// ---------------- problem constants (from reference) ----------------
static constexpr int NPn = 10000;   // pitchers
static constexpr int NBn = 30000;   // batters
static constexpr int En  = 300000;  // edges
static constexpr int BEn = 8192;    // batch edges
static constexpr int HIDc = 64, GHc = 4, GHID = 256; // GH*HID
static constexpr int Dc = 128, SEQc = 6, NCc = 10;

typedef __attribute__((ext_vector_type(16))) _Float16 v16h;
typedef __attribute__((ext_vector_type(4)))  _Float16 v4h;
typedef __attribute__((ext_vector_type(8)))  float    v8f;

__device__ __forceinline__ float gelu_f(float x) {
    return 0.5f * x * (1.0f + erff(x * 0.70710678118654752f));
}
// monotonic float<->uint mapping for atomicMax on floats
__device__ __forceinline__ unsigned fmap(float f) {
    unsigned u = __float_as_uint(f);
    return (u & 0x80000000u) ? ~u : (u | 0x80000000u);
}
__device__ __forceinline__ float funmap(unsigned u) {
    return (u & 0x80000000u) ? __uint_as_float(u & 0x7FFFFFFFu) : __uint_as_float(~u);
}
static constexpr unsigned MAPPED_NEG_INF = 0x007FFFFFu; // fmap(-inf)

// ---------------- generic WMMA GEMM: C = act(A[M,K] @ B[K,N] + bias) -------
// wave = 16x64 C strip (4 x v_wmma_f32_16x16x32_f16 per K-chunk, one shared A
// fragment); block = 256 thr = 8 waves stacked on M; grid = (N/64, ceil(M/128)).
// A fragment: 4x b128 global loads per lane (one contiguous 16-float run).
// B chunk 32x64 is staged in LDS *pre-swizzled into the WMMA B-fragment
// layout*: flat index ((sub*32 + lane)*16 + e) holds
// B[k0 + (lane>=16 ? 16 : 0) + e][n0 + sub*16 + (lane&15)], so each lane's
// fragment is one contiguous 32B run -> 2x ds_load_b128, no packing movs.
__global__ __launch_bounds__(256) void gemm_wmma_f16(
    const float* __restrict__ A, const float* __restrict__ B,
    const float* __restrict__ bias, float* __restrict__ C,
    int M, int N, int K, int act)
{
    __shared__ __align__(32) _Float16 Bf[4 * 32 * 16];   // 4KB, fragment layout
    const int lane = threadIdx.x & 31;
    const int wave = threadIdx.x >> 5;
    const int n0 = blockIdx.x * 64;
    const int m0 = (blockIdx.y * 8 + wave) * 16;
    const int mrow  = lane & 15;
    const int hi    = lane >> 4;        // 0 or 1
    const int abase = hi << 4;          // A frag K offset 0 / 16
    const int m = m0 + mrow;

    v8f acc[4] = {};

    for (int k0 = 0; k0 < K; k0 += 32) {
        const bool fullK = (k0 + 32 <= K);
        // ---- stage B chunk (32 x 64) into LDS in fragment layout ----
        if (fullK && n0 + 64 <= N) {
            const int t  = threadIdx.x;
            const int nn = t & 63;            // column within the 64-wide strip
            const int r0 = (t >> 6) * 8;      // 8-row K-run: 0,8,16,24
            const float* bp = B + (size_t)(k0 + r0) * N + n0 + nn;
            const int sub = nn >> 4;
            const int bl  = (nn & 15) + (r0 >= 16 ? 16 : 0);
            _Float16* dst = &Bf[(sub * 32 + bl) * 16 + (r0 & 15)];
            v4h h0, h1;
            h0[0] = (_Float16)bp[0 * (size_t)N]; h0[1] = (_Float16)bp[1 * (size_t)N];
            h0[2] = (_Float16)bp[2 * (size_t)N]; h0[3] = (_Float16)bp[3 * (size_t)N];
            h1[0] = (_Float16)bp[4 * (size_t)N]; h1[1] = (_Float16)bp[5 * (size_t)N];
            h1[2] = (_Float16)bp[6 * (size_t)N]; h1[3] = (_Float16)bp[7 * (size_t)N];
            *(v4h*)dst = h0;
            *(v4h*)(dst + 4) = h1;
        } else {
            for (int t2 = threadIdx.x; t2 < 2048; t2 += 256) {
                int kk = t2 >> 6, nn2 = t2 & 63;
                float v = 0.0f;
                if (k0 + kk < K && n0 + nn2 < N) v = B[(size_t)(k0 + kk) * N + (n0 + nn2)];
                int sub = nn2 >> 4;
                int bl  = (nn2 & 15) + (kk >= 16 ? 16 : 0);
                Bf[(sub * 32 + bl) * 16 + (kk & 15)] = (_Float16)v;
            }
        }
        __syncthreads();
        // ---- A fragment (16-bit 16x32, contiguous k = 16*hi + e, same order as Bf) ----
        v16h af;
        if (m < M && fullK) {
            const float* ap = A + (size_t)m * K + k0 + abase;
            float4 a0 = *(const float4*)(ap);
            float4 a1 = *(const float4*)(ap + 4);
            float4 a2 = *(const float4*)(ap + 8);
            float4 a3 = *(const float4*)(ap + 12);
            af[0]  = (_Float16)a0.x; af[1]  = (_Float16)a0.y;
            af[2]  = (_Float16)a0.z; af[3]  = (_Float16)a0.w;
            af[4]  = (_Float16)a1.x; af[5]  = (_Float16)a1.y;
            af[6]  = (_Float16)a1.z; af[7]  = (_Float16)a1.w;
            af[8]  = (_Float16)a2.x; af[9]  = (_Float16)a2.y;
            af[10] = (_Float16)a2.z; af[11] = (_Float16)a2.w;
            af[12] = (_Float16)a3.x; af[13] = (_Float16)a3.y;
            af[14] = (_Float16)a3.z; af[15] = (_Float16)a3.w;
            if (k0 + 64 < K) __builtin_prefetch(ap + 32, 0, 1);
        } else {
            #pragma unroll
            for (int e = 0; e < 16; ++e) {
                int ka = abase + e;
                float av = 0.0f;
                if (m < M && k0 + ka < K) av = A[(size_t)m * K + k0 + ka];
                af[e] = (_Float16)av;
            }
        }
        // ---- 4 WMMAs over the 16-wide N subtiles ----
        #pragma unroll
        for (int sub = 0; sub < 4; ++sub) {
            v16h bf = *(const v16h*)(&Bf[(sub * 32 + lane) * 16]);
            acc[sub] = __builtin_amdgcn_wmma_f32_16x16x32_f16(
                false, af, false, bf, (short)0, acc[sub], false, false);
        }
        __syncthreads();
    }
    // ---- store ----
    #pragma unroll
    for (int sub = 0; sub < 4; ++sub) {
        int n = n0 + sub * 16 + mrow;
        if (n < N) {
            float bv = bias ? bias[n] : 0.0f;
            #pragma unroll
            for (int r = 0; r < 8; ++r) {
                int mr = m0 + r + (hi << 3);
                if (mr < M) {
                    float v = acc[sub][r] + bv;
                    if (act == 1) v = gelu_f(v);
                    C[(size_t)mr * N + n] = v;
                }
            }
        }
    }
}

// ---------------- fills ----------------
__global__ void fill_u32_k(unsigned* p, unsigned v, int n) {
    int i = blockIdx.x * blockDim.x + threadIdx.x;
    if (i < n) p[i] = v;
}
__global__ void fill_f32_k(float* p, float v, int n) {
    int i = blockIdx.x * blockDim.x + threadIdx.x;
    if (i < n) p[i] = v;
}

// ---------------- GATv2 edge passes ----------------
// pass 1: score[e,h] = sum_c lrelu(xl[s,h,c]+xr[d,h,c]+(ea[e]@We)[h,c]) * att[h,c]
__global__ __launch_bounds__(256) void gat_edge_score(
    const float* __restrict__ xl, const float* __restrict__ xr,
    const float* __restrict__ ea, const float* __restrict__ We,
    const float* __restrict__ att,
    const int* __restrict__ src, const int* __restrict__ dst,
    float* __restrict__ score, unsigned* __restrict__ segmax, int E)
{
    __shared__ float sWe[16 * GHID];
    __shared__ float sAtt[GHc * HIDc];
    for (int t = threadIdx.x; t < 16 * GHID; t += blockDim.x) sWe[t] = We[t];
    for (int t = threadIdx.x; t < GHc * HIDc; t += blockDim.x) sAtt[t] = att[t];
    __syncthreads();
    int idx = blockIdx.x * blockDim.x + threadIdx.x;
    if (idx >= E * GHc) return;
    int e = idx >> 2, h = idx & 3;
    int s = src[e], d = dst[e];
    float4 ev0 = *(const float4*)(ea + (size_t)e * 16);
    float4 ev1 = *(const float4*)(ea + (size_t)e * 16 + 4);
    float4 ev2 = *(const float4*)(ea + (size_t)e * 16 + 8);
    float4 ev3 = *(const float4*)(ea + (size_t)e * 16 + 12);
    float eav[16] = {ev0.x, ev0.y, ev0.z, ev0.w, ev1.x, ev1.y, ev1.z, ev1.w,
                     ev2.x, ev2.y, ev2.z, ev2.w, ev3.x, ev3.y, ev3.z, ev3.w};
    const float* xls = xl + (size_t)s * GHID + h * HIDc;
    const float* xrd = xr + (size_t)d * GHID + h * HIDc;
    const float* we  = sWe + h * HIDc;           // sWe[j*GHID + h*64 + c]
    const float* at  = sAtt + h * HIDc;
    float sc = 0.0f;
    for (int c = 0; c < HIDc; c += 4) {
        float4 xa = *(const float4*)(xls + c);
        float4 xb = *(const float4*)(xrd + c);
        float eh[4] = {0.0f, 0.0f, 0.0f, 0.0f};
        #pragma unroll
        for (int j = 0; j < 16; ++j) {
            const float* w = we + j * GHID + c;
            eh[0] += eav[j] * w[0]; eh[1] += eav[j] * w[1];
            eh[2] += eav[j] * w[2]; eh[3] += eav[j] * w[3];
        }
        float m0v = xa.x + xb.x + eh[0], m1v = xa.y + xb.y + eh[1];
        float m2v = xa.z + xb.z + eh[2], m3v = xa.w + xb.w + eh[3];
        m0v = (m0v > 0.0f) ? m0v : 0.2f * m0v;
        m1v = (m1v > 0.0f) ? m1v : 0.2f * m1v;
        m2v = (m2v > 0.0f) ? m2v : 0.2f * m2v;
        m3v = (m3v > 0.0f) ? m3v : 0.2f * m3v;
        sc += m0v * at[c] + m1v * at[c + 1] + m2v * at[c + 2] + m3v * at[c + 3];
    }
    score[idx] = sc;
    atomicMax(&segmax[d * GHc + h], fmap(sc));
}

// pass 2: ex = exp(score - segmax); segsum += ex (score overwritten with ex)
__global__ __launch_bounds__(256) void gat_edge_exp(
    const int* __restrict__ dst, float* __restrict__ score,
    const unsigned* __restrict__ segmax, float* __restrict__ segsum, int E)
{
    int idx = blockIdx.x * blockDim.x + threadIdx.x;
    if (idx >= E * GHc) return;
    int e = idx >> 2, h = idx & 3;
    int d = dst[e];
    float m = funmap(segmax[d * GHc + h]);
    if (!isfinite(m)) m = 0.0f;
    float ex = expf(score[idx] - m);
    score[idx] = ex;
    atomicAdd(&segsum[d * GHc + h], ex);
}

// pass 3: agg[d,h,:] += alpha * xl[s,h,:]
__global__ __launch_bounds__(256) void gat_edge_scatter(
    const float* __restrict__ xl, const int* __restrict__ src,
    const int* __restrict__ dst, const float* __restrict__ score,
    const float* __restrict__ segsum, float* __restrict__ agg, int E)
{
    int idx = blockIdx.x * blockDim.x + threadIdx.x;
    if (idx >= E * GHc) return;
    int e = idx >> 2, h = idx & 3;
    int s = src[e], d = dst[e];
    float alpha = score[idx] / (segsum[d * GHc + h] + 1e-16f);
    const float* xls = xl + (size_t)s * GHID + h * HIDc;
    float* ag = agg + (size_t)d * GHID + h * HIDc;
    for (int c = 0; c < HIDc; c += 4) {
        float4 xa = *(const float4*)(xls + c);
        atomicAdd(&ag[c],     alpha * xa.x);
        atomicAdd(&ag[c + 1], alpha * xa.y);
        atomicAdd(&ag[c + 2], alpha * xa.z);
        atomicAdd(&ag[c + 3], alpha * xa.w);
    }
}

// node epilogue: h = gelu(mean_heads(agg) + bias), vectorized over 4 channels
__global__ void gat_finalize(const float* __restrict__ agg,
                             const float* __restrict__ bias,
                             float* __restrict__ hout, int Nd)
{
    int idx = blockIdx.x * blockDim.x + threadIdx.x;
    if (idx >= Nd * 16) return;
    int d = idx >> 4, c = (idx & 15) * 4;
    const float* a = agg + (size_t)d * GHID;
    float4 h0 = *(const float4*)(a + c);
    float4 h1 = *(const float4*)(a + HIDc + c);
    float4 h2 = *(const float4*)(a + 2 * HIDc + c);
    float4 h3 = *(const float4*)(a + 3 * HIDc + c);
    float4 bv = *(const float4*)(bias + c);
    float4 o;
    o.x = gelu_f(0.25f * (h0.x + h1.x + h2.x + h3.x) + bv.x);
    o.y = gelu_f(0.25f * (h0.y + h1.y + h2.y + h3.y) + bv.y);
    o.z = gelu_f(0.25f * (h0.z + h1.z + h2.z + h3.z) + bv.z);
    o.w = gelu_f(0.25f * (h0.w + h1.w + h2.w + h3.w) + bv.w);
    *(float4*)(hout + (size_t)d * HIDc + c) = o;
}

// ---------------- column mean: out[c] = mean_r A[r,c] ----------------
__global__ __launch_bounds__(256) void col_mean(const float* __restrict__ A,
                                                int M, int N, float* __restrict__ out)
{
    int c = blockIdx.x;
    float s = 0.0f;
    for (int r = threadIdx.x; r < M; r += blockDim.x) s += A[(size_t)r * N + c];
    __shared__ float red[256];
    red[threadIdx.x] = s;
    __syncthreads();
    for (int off = 128; off > 0; off >>= 1) {
        if (threadIdx.x < off) red[threadIdx.x] += red[threadIdx.x + off];
        __syncthreads();
    }
    if (threadIdx.x == 0) out[c] = red[0] / (float)M;
}

// state = gsin[144] @ Wgs[144,128] + bgs
__global__ void matvec144(const float* __restrict__ g, const float* __restrict__ W,
                          const float* __restrict__ b, float* __restrict__ out)
{
    int c = blockIdx.x * blockDim.x + threadIdx.x;
    if (c >= Dc) return;
    float s = b[c];
    for (int j = 0; j < 144; ++j) s += g[j] * W[j * Dc + c];
    out[c] = s;
}

// gather token inputs G[i,0:144] = [h_p[src[eid]], h_b[dst[eid]], ea[eid]]
__global__ void tok_gather(const float* __restrict__ hp, const float* __restrict__ hb,
                           const float* __restrict__ ea, const int* __restrict__ src,
                           const int* __restrict__ dst, const int* __restrict__ eids,
                           float* __restrict__ G)
{
    int idx = blockIdx.x * blockDim.x + threadIdx.x;  // BE * 36 quads
    if (idx >= BEn * 36) return;
    int i = idx / 36, q = idx % 36;
    int e = eids[i];
    float4 v;
    if (q < 16)      v = *(const float4*)(hp + (size_t)src[e] * HIDc + q * 4);
    else if (q < 32) v = *(const float4*)(hb + (size_t)dst[e] * HIDc + (q - 16) * 4);
    else             v = *(const float4*)(ea + (size_t)e * 16 + (q - 32) * 4);
    *(float4*)(G + (size_t)i * 144 + q * 4) = v;
}

// ---------------- tiny transformer (6 tokens x 128), single block -----------
__global__ __launch_bounds__(128) void transformer_k(
    const float* __restrict__ hist, const float* __restrict__ state,
    const float* __restrict__ pos,
    const float* __restrict__ Wqkv, const float* __restrict__ bqkv,
    const float* __restrict__ Wo,   const float* __restrict__ bo,
    const float* __restrict__ ln1g, const float* __restrict__ ln1b,
    const float* __restrict__ W1,   const float* __restrict__ b1,
    const float* __restrict__ W2,   const float* __restrict__ b2,
    const float* __restrict__ ln2g, const float* __restrict__ ln2b,
    float* __restrict__ ctx)
{
    __shared__ float x[SEQc][Dc], q[SEQc][Dc], kk_[SEQc][Dc], v[SEQc][Dc];
    __shared__ float o[SEQc][Dc], tmp[SEQc][Dc], ff[SEQc][4 * Dc];
    __shared__ float attn[4][SEQc][SEQc];
    __shared__ float mu[SEQc], rs[SEQc];
    int t = threadIdx.x;
    for (int s = 0; s < SEQc; ++s) {
        float hv = (s < SEQc - 1) ? hist[s * Dc + t] : state[t];
        x[s][t] = hv + pos[s * Dc + t];
    }
    __syncthreads();
    for (int l = 0; l < 2; ++l) {
        const float* Wq = Wqkv + (size_t)l * Dc * 3 * Dc;
        const float* bq = bqkv + l * 3 * Dc;
        for (int s = 0; s < SEQc; ++s)
            for (int p = 0; p < 3; ++p) {
                int col = t + p * Dc;
                float a = bq[col];
                for (int u = 0; u < Dc; ++u) a += x[s][u] * Wq[u * 3 * Dc + col];
                if (p == 0) q[s][t] = a; else if (p == 1) kk_[s][t] = a; else v[s][t] = a;
            }
        __syncthreads();
        if (t < 4 * SEQc * SEQc) {
            int h = t / 36, rem = t % 36, i = rem / 6, j = rem % 6;
            float s = 0.0f;
            for (int d2 = 0; d2 < 32; ++d2) s += q[i][h * 32 + d2] * kk_[j][h * 32 + d2];
            attn[h][i][j] = s * 0.17677669529663687f;   // 1/sqrt(32)
        }
        __syncthreads();
        if (t < 4 * SEQc) {
            int h = t / 6, i = t % 6;
            float mx = -1e30f;
            for (int j = 0; j < SEQc; ++j) mx = fmaxf(mx, attn[h][i][j]);
            float s = 0.0f;
            for (int j = 0; j < SEQc; ++j) { float e = expf(attn[h][i][j] - mx); attn[h][i][j] = e; s += e; }
            for (int j = 0; j < SEQc; ++j) attn[h][i][j] /= s;
        }
        __syncthreads();
        for (int s = 0; s < SEQc; ++s) {
            int h = t >> 5;
            float a = 0.0f;
            for (int j = 0; j < SEQc; ++j) a += attn[h][s][j] * v[j][t];
            o[s][t] = a;
        }
        __syncthreads();
        const float* Wol = Wo + (size_t)l * Dc * Dc; const float* bol = bo + l * Dc;
        for (int s = 0; s < SEQc; ++s) {
            float a = bol[t];
            for (int u = 0; u < Dc; ++u) a += o[s][u] * Wol[u * Dc + t];
            tmp[s][t] = x[s][t] + a;
        }
        __syncthreads();
        if (t < SEQc) {
            float m = 0.0f; for (int u = 0; u < Dc; ++u) m += tmp[t][u]; m /= Dc;
            float vv = 0.0f; for (int u = 0; u < Dc; ++u) { float d = tmp[t][u] - m; vv += d * d; }
            mu[t] = m; rs[t] = rsqrtf(vv / Dc + 1e-5f);
        }
        __syncthreads();
        for (int s = 0; s < SEQc; ++s)
            x[s][t] = (tmp[s][t] - mu[s]) * rs[s] * ln1g[l * Dc + t] + ln1b[l * Dc + t];
        __syncthreads();
        const float* W1l = W1 + (size_t)l * Dc * 4 * Dc; const float* b1l = b1 + l * 4 * Dc;
        for (int p = 0; p < 4; ++p) {
            int col = t + p * Dc;
            for (int s = 0; s < SEQc; ++s) {
                float a = b1l[col];
                for (int u = 0; u < Dc; ++u) a += x[s][u] * W1l[u * 4 * Dc + col];
                ff[s][col] = gelu_f(a);
            }
        }
        __syncthreads();
        const float* W2l = W2 + (size_t)l * 4 * Dc * Dc; const float* b2l = b2 + l * Dc;
        for (int s = 0; s < SEQc; ++s) {
            float a = b2l[t];
            for (int u = 0; u < 4 * Dc; ++u) a += ff[s][u] * W2l[u * Dc + t];
            tmp[s][t] = x[s][t] + a;
        }
        __syncthreads();
        if (t < SEQc) {
            float m = 0.0f; for (int u = 0; u < Dc; ++u) m += tmp[t][u]; m /= Dc;
            float vv = 0.0f; for (int u = 0; u < Dc; ++u) { float d = tmp[t][u] - m; vv += d * d; }
            mu[t] = m; rs[t] = rsqrtf(vv / Dc + 1e-5f);
        }
        __syncthreads();
        for (int s = 0; s < SEQc; ++s)
            x[s][t] = (tmp[s][t] - mu[s]) * rs[s] * ln2g[l * Dc + t] + ln2b[l * Dc + t];
        __syncthreads();
    }
    ctx[t] = x[SEQc - 1][t];
}

// cat[i, 0:128] = tok[i]; cat[i, 128:256] = ctx   (float4 per thread)
__global__ void build_cat(const float* __restrict__ tok, const float* __restrict__ ctx,
                          float* __restrict__ cat)
{
    int idx = blockIdx.x * blockDim.x + threadIdx.x;  // BE * 64 quads
    if (idx >= BEn * 64) return;
    int i = idx >> 6, q = idx & 63;
    float4 v = (q < 32) ? *(const float4*)(tok + (size_t)i * Dc + q * 4)
                        : *(const float4*)(ctx + (q - 32) * 4);
    *(float4*)(cat + (size_t)i * 256 + q * 4) = v;
}

// out[i,c] = hid1[i,:] @ Wc2[:,c] + bc2[c]
__global__ void head2_k(const float* __restrict__ H, const float* __restrict__ Wc2,
                        const float* __restrict__ bc2, float* __restrict__ out)
{
    int idx = blockIdx.x * blockDim.x + threadIdx.x;
    if (idx >= BEn * NCc) return;
    int i = idx / NCc, c = idx % NCc;
    float s = bc2[c];
    const float4* h4 = (const float4*)(H + (size_t)i * Dc);
    for (int u = 0; u < Dc / 4; ++u) {
        float4 hv = h4[u];
        s += hv.x * Wc2[(4 * u) * NCc + c]     + hv.y * Wc2[(4 * u + 1) * NCc + c]
           + hv.z * Wc2[(4 * u + 2) * NCc + c] + hv.w * Wc2[(4 * u + 3) * NCc + c];
    }
    out[idx] = s;
}

// ---------------- host launcher ----------------
extern "C" void kernel_launch(void* const* d_in, const int* in_sizes, int n_in,
                              void* d_out, int out_size, void* d_ws, size_t ws_size,
                              hipStream_t stream)
{
    (void)in_sizes; (void)n_in; (void)ws_size; (void)out_size;
    const float* pitcher = (const float*)d_in[0];
    const float* batter  = (const float*)d_in[1];
    const float* ea      = (const float*)d_in[2];
    const float* hist    = (const float*)d_in[3];
    const float* Wp = (const float*)d_in[4];  const float* bp = (const float*)d_in[5];
    const float* Wb = (const float*)d_in[6];  const float* bb = (const float*)d_in[7];
    const float* rel_Wl = (const float*)d_in[8];  const float* rel_bl = (const float*)d_in[9];
    const float* rel_Wr = (const float*)d_in[10]; const float* rel_br = (const float*)d_in[11];
    const float* rel_We = (const float*)d_in[12]; const float* rel_att = (const float*)d_in[13];
    const float* rel_bias = (const float*)d_in[14];
    const float* rev_Wl = (const float*)d_in[15]; const float* rev_bl = (const float*)d_in[16];
    const float* rev_Wr = (const float*)d_in[17]; const float* rev_br = (const float*)d_in[18];
    const float* rev_We = (const float*)d_in[19]; const float* rev_att = (const float*)d_in[20];
    const float* rev_bias = (const float*)d_in[21];
    const float* Wgs = (const float*)d_in[22]; const float* bgs = (const float*)d_in[23];
    const float* Wet = (const float*)d_in[24]; const float* bet = (const float*)d_in[25];
    const float* pos = (const float*)d_in[26];
    const float* Wqkv = (const float*)d_in[27]; const float* bqkv = (const float*)d_in[28];
    const float* Wo = (const float*)d_in[29];   const float* bo = (const float*)d_in[30];
    const float* ln1g = (const float*)d_in[31]; const float* ln1b = (const float*)d_in[32];
    const float* W1 = (const float*)d_in[33];   const float* b1 = (const float*)d_in[34];
    const float* W2 = (const float*)d_in[35];   const float* b2 = (const float*)d_in[36];
    const float* ln2g = (const float*)d_in[37]; const float* ln2b = (const float*)d_in[38];
    const float* Wc1 = (const float*)d_in[39];  const float* bc1 = (const float*)d_in[40];
    const float* Wc2 = (const float*)d_in[41];  const float* bc2 = (const float*)d_in[42];
    const int* esrc = (const int*)d_in[43];
    const int* edst = (const int*)d_in[44];
    const int* eids = (const int*)d_in[45];

    float* ws = (float*)d_ws;
    size_t off = 0;
    auto alloc = [&](size_t nelem) -> float* {
        float* p = ws + off;
        off += (nelem + 255) & ~(size_t)255;
        return p;
    };
    // persistent region
    float* px   = alloc((size_t)NPn * HIDc);
    float* bx   = alloc((size_t)NBn * HIDc);
    float* h_p  = alloc((size_t)NPn * HIDc);
    float* h_b  = alloc((size_t)NBn * HIDc);
    float* gsin = alloc(256);
    float* stateB = alloc(256);
    float* ctxB   = alloc(256);
    size_t reuse_base = off;
    // GAT scratch region (recycled later)
    float* xlrel = alloc((size_t)NPn * GHID);
    float* xrrel = alloc((size_t)NBn * GHID);
    float* xlrev = alloc((size_t)NBn * GHID);
    float* xrrev = alloc((size_t)NPn * GHID);
    float* sc_rel = alloc((size_t)En * GHc);
    float* sc_rev = alloc((size_t)En * GHc);
    unsigned* smax_rel = (unsigned*)alloc((size_t)NBn * GHc);
    float*    ssum_rel = alloc((size_t)NBn * GHc);
    unsigned* smax_rev = (unsigned*)alloc((size_t)NPn * GHc);
    float*    ssum_rev = alloc((size_t)NPn * GHc);
    float* agg_rel = alloc((size_t)NBn * GHID);
    float* agg_rev = alloc((size_t)NPn * GHID);

    const int T = 256;
    auto cdiv = [](int a, int b) { return (a + b - 1) / b; };

    // node encoders (WMMA, gelu)
    gemm_wmma_f16<<<dim3(1, cdiv(NPn, 128)), T, 0, stream>>>(pitcher, Wp, bp, px, NPn, HIDc, 128, 1);
    gemm_wmma_f16<<<dim3(1, cdiv(NBn, 128)), T, 0, stream>>>(batter,  Wb, bb, bx, NBn, HIDc, 128, 1);
    // GATv2 linear transforms (WMMA)
    gemm_wmma_f16<<<dim3(4, cdiv(NPn, 128)), T, 0, stream>>>(px, rel_Wl, rel_bl, xlrel, NPn, GHID, HIDc, 0);
    gemm_wmma_f16<<<dim3(4, cdiv(NBn, 128)), T, 0, stream>>>(bx, rel_Wr, rel_br, xrrel, NBn, GHID, HIDc, 0);
    gemm_wmma_f16<<<dim3(4, cdiv(NBn, 128)), T, 0, stream>>>(bx, rev_Wl, rev_bl, xlrev, NBn, GHID, HIDc, 0);
    gemm_wmma_f16<<<dim3(4, cdiv(NPn, 128)), T, 0, stream>>>(px, rev_Wr, rev_br, xrrev, NPn, GHID, HIDc, 0);
    // init accumulators
    fill_u32_k<<<cdiv(NBn * GHc, T), T, 0, stream>>>(smax_rel, MAPPED_NEG_INF, NBn * GHc);
    fill_u32_k<<<cdiv(NPn * GHc, T), T, 0, stream>>>(smax_rev, MAPPED_NEG_INF, NPn * GHc);
    fill_f32_k<<<cdiv(NBn * GHc, T), T, 0, stream>>>(ssum_rel, 0.0f, NBn * GHc);
    fill_f32_k<<<cdiv(NPn * GHc, T), T, 0, stream>>>(ssum_rev, 0.0f, NPn * GHc);
    fill_f32_k<<<cdiv(NBn * GHID, T), T, 0, stream>>>(agg_rel, 0.0f, NBn * GHID);
    fill_f32_k<<<cdiv(NPn * GHID, T), T, 0, stream>>>(agg_rev, 0.0f, NPn * GHID);
    // edge passes (rel: pitcher->batter, seg by edge_dst; rev: batter->pitcher, seg by edge_src)
    int eg = cdiv(En * GHc, T);
    gat_edge_score<<<eg, T, 0, stream>>>(xlrel, xrrel, ea, rel_We, rel_att, esrc, edst, sc_rel, smax_rel, En);
    gat_edge_score<<<eg, T, 0, stream>>>(xlrev, xrrev, ea, rev_We, rev_att, edst, esrc, sc_rev, smax_rev, En);
    gat_edge_exp<<<eg, T, 0, stream>>>(edst, sc_rel, smax_rel, ssum_rel, En);
    gat_edge_exp<<<eg, T, 0, stream>>>(esrc, sc_rev, smax_rev, ssum_rev, En);
    gat_edge_scatter<<<eg, T, 0, stream>>>(xlrel, esrc, edst, sc_rel, ssum_rel, agg_rel, En);
    gat_edge_scatter<<<eg, T, 0, stream>>>(xlrev, edst, esrc, sc_rev, ssum_rev, agg_rev, En);
    gat_finalize<<<cdiv(NBn * 16, T), T, 0, stream>>>(agg_rel, rel_bias, h_b, NBn);
    gat_finalize<<<cdiv(NPn * 16, T), T, 0, stream>>>(agg_rev, rev_bias, h_p, NPn);
    // global state vector
    col_mean<<<HIDc, T, 0, stream>>>(h_p, NPn, HIDc, gsin);
    col_mean<<<HIDc, T, 0, stream>>>(h_b, NBn, HIDc, gsin + 64);
    col_mean<<<16,   T, 0, stream>>>(ea, En, 16, gsin + 128);
    matvec144<<<1, Dc, 0, stream>>>(gsin, Wgs, bgs, stateB);
    // transformer over [hist; state]
    transformer_k<<<1, Dc, 0, stream>>>(hist, stateB, pos, Wqkv, bqkv, Wo, bo,
                                        ln1g, ln1b, W1, b1, W2, b2, ln2g, ln2b, ctxB);
    // reuse GAT scratch region for token/head buffers
    off = reuse_base;
    float* G    = alloc((size_t)BEn * 144);
    float* tok  = alloc((size_t)BEn * Dc);
    float* cat  = alloc((size_t)BEn * 2 * Dc);
    float* hid1 = alloc((size_t)BEn * Dc);

    tok_gather<<<cdiv(BEn * 36, T), T, 0, stream>>>(h_p, h_b, ea, esrc, edst, eids, G);
    gemm_wmma_f16<<<dim3(2, cdiv(BEn, 128)), T, 0, stream>>>(G, Wet, bet, tok, BEn, Dc, 144, 0);
    build_cat<<<cdiv(BEn * 64, T), T, 0, stream>>>(tok, ctxB, cat);
    gemm_wmma_f16<<<dim3(2, cdiv(BEn, 128)), T, 0, stream>>>(cat, Wc1, bc1, hid1, BEn, Dc, 2 * Dc, 1);
    head2_k<<<cdiv(BEn * NCc, T), T, 0, stream>>>(hid1, Wc2, bc2, (float*)d_out);
}
